// TransformerBlock_34325378629935
// MI455X (gfx1250) — matrix operand order repaired and checked by reading, not compiled, not run on hardware
//
#include <hip/hip_runtime.h>
#include <hip/hip_bf16.h>

typedef __bf16 bf16;
typedef __attribute__((ext_vector_type(16))) __bf16 v16bf;
typedef __attribute__((ext_vector_type(8)))  __bf16 v8bf;
typedef __attribute__((ext_vector_type(8)))  float  v8f;
typedef __attribute__((ext_vector_type(4)))  unsigned int u32x4;
typedef __attribute__((ext_vector_type(8)))  int i32x8;
typedef __attribute__((ext_vector_type(4)))  int i32x4;

#define DIM   48
#define C3    144
#define HIDN  192
#define NHEAD 8
#define HC    6
#define BATCH 4
#define HH    256
#define WW    256
#define NPIX  65536
#define TN    128

#if __has_builtin(__builtin_amdgcn_tensor_load_to_lds) && \
    __has_builtin(__builtin_amdgcn_s_wait_tensorcnt)
#define USE_TDM 1
#endif

static __device__ inline bf16 f2bf(float v) { return (bf16)v; }

#ifdef USE_TDM
// Issue a 2D TDM tile load: tile_d1 rows x tile_d0 elems (elemBytes each),
// row stride = strideElems, from global tile start -> contiguous LDS tile.
// Must be executed by exactly one wave; follow with s_wait_tensorcnt + barrier.
static __device__ inline void tdm_load_2d(const void* gptr, unsigned int lds_addr,
                                          unsigned int tile_d0, unsigned int tile_d1,
                                          unsigned long long tensor_d0,
                                          unsigned long long tensor_d1,
                                          unsigned long long strideElems,
                                          unsigned int dataSizeCode) {
    unsigned long long ga = (unsigned long long)(uintptr_t)gptr;
    u32x4 g0;
    g0[0] = 1u;                                       // count=1, user descriptor
    g0[1] = lds_addr;                                 // LDS byte address
    g0[2] = (unsigned int)(ga & 0xFFFFFFFFu);         // global_addr[31:0]
    g0[3] = (unsigned int)((ga >> 32) & 0x01FFFFFFu)  // global_addr[56:32]
            | (2u << 30);                             // type=2 ("image")
    i32x8 g1;
    g1[0] = (int)(dataSizeCode << 16);                // mask=0, data_size
    g1[1] = (int)((tensor_d0 & 0xFFFFu) << 16);       // dim0[15:0]
    g1[2] = (int)(((tensor_d0 >> 16) & 0xFFFFu) | ((tensor_d1 & 0xFFFFu) << 16));
    g1[3] = (int)(((tensor_d1 >> 16) & 0xFFFFu) | (tile_d0 << 16));
    g1[4] = (int)(tile_d1 & 0xFFFFu);                 // tile_dim1, tile_dim2=0
    g1[5] = (int)(strideElems & 0xFFFFFFFFull);       // dim0_stride[31:0]
    g1[6] = (int)((strideElems >> 32) & 0xFFFFull);   // dim0_stride[47:32]
    g1[7] = 0;
    i32x4 z4 = {0, 0, 0, 0};
#if __clang_major__ >= 23
    i32x8 z8 = {0, 0, 0, 0, 0, 0, 0, 0};
    __builtin_amdgcn_tensor_load_to_lds(g0, g1, z4, z4, z8, 0);
#else
    __builtin_amdgcn_tensor_load_to_lds(g0, g1, z4, z4, 0);
#endif
}
#endif

// ---- WMMA fragment loaders (contiguous k = 16*half + s, wave32) -------------------
static __device__ inline v16bf load_a_frag(const bf16* __restrict__ W, int Kpad,
                                           int mt, int ks, int lane) {
    int half = lane >> 4, m = lane & 15;
    const bf16* rp = W + (size_t)(mt * 16 + m) * Kpad + ks * 32 + 16 * half;
    const uint32_t* rp32 = reinterpret_cast<const uint32_t*>(rp);
    v16bf a;
    uint32_t* ap = reinterpret_cast<uint32_t*>(&a);
#pragma unroll
    for (int v = 0; v < 4; v++) ap[v] = rp32[v];
#pragma unroll
    for (int v = 0; v < 4; v++) ap[4 + v] = rp32[4 + v];
    return a;
}

static __device__ inline v16bf load_b_frag(const bf16* __restrict__ pk, int strideK,
                                           int colBase, int ks, int lane) {
    int half = lane >> 4, col = lane & 15;
    const bf16* p = pk + (size_t)(colBase + col) * strideK + ks * 32 + 16 * half;
    v16bf r;
    v8bf* rp = reinterpret_cast<v8bf*>(&r);
    rp[0] = *reinterpret_cast<const v8bf*>(p);
    rp[1] = *reinterpret_cast<const v8bf*>(p + 8);
    return r;
}

// ---- weight prep: fp32 -> bf16, K padded to 64 ----------------------------
__global__ __launch_bounds__(256) void k_prep(const float* __restrict__ wqkv,
                                              const float* __restrict__ wf1,
                                              const float* __restrict__ wf2,
                                              bf16* __restrict__ wqkvb,
                                              bf16* __restrict__ wf1b,
                                              bf16* __restrict__ wf2b) {
    int e = blockIdx.x * 256 + threadIdx.x;
    if (e < C3 * 64) {
        int o = e >> 6, kk = e & 63;
        wqkvb[e] = f2bf(kk < DIM ? wqkv[o * DIM + kk] : 0.f);
        return;
    }
    e -= C3 * 64;
    if (e < HIDN * 64) {
        int o = e >> 6, kk = e & 63;
        wf1b[e] = f2bf(kk < DIM ? wf1[o * DIM + kk] : 0.f);
        return;
    }
    e -= HIDN * 64;
    if (e < DIM * HIDN) wf2b[e] = f2bf(wf2[e]);
}

// ---- fused LN1 + qkv GEMM (M=144, K=48->64) -------------------------------
__global__ __launch_bounds__(256) void k_qkv(const float* __restrict__ x,
                                             const float* __restrict__ g,
                                             const float* __restrict__ bln,
                                             const bf16* __restrict__ Wb,
                                             const float* __restrict__ bias,
                                             bf16* __restrict__ out) {
    __shared__ __align__(32) float xs[DIM][TN];
    __shared__ __align__(32) bf16 ypk[TN * 72];
    int tid = threadIdx.x;
    int pix0 = blockIdx.x * TN;
    int b = blockIdx.y;
    const float* xb = x + (size_t)b * DIM * NPIX;
#ifdef USE_TDM
    // TDM DMA: 48x128 fp32 tile (row stride NPIX) -> contiguous LDS xs[48][128]
    if (tid < 32) {
        tdm_load_2d(xb + pix0, (unsigned int)(uintptr_t)&xs[0][0],
                    TN, DIM, NPIX, DIM, NPIX, 2u /*4-byte elems*/);
        __builtin_amdgcn_s_wait_tensorcnt(0);
    }
#else
    for (int e = tid; e < DIM * TN; e += 256) {
        int c = e >> 7, i = e & 127;
        xs[c][i] = xb[(size_t)c * NPIX + pix0 + i];
    }
#endif
    __syncthreads();
    if (tid < TN) {
        float s = 0.f, s2 = 0.f;
#pragma unroll
        for (int c = 0; c < DIM; c++) { float v = xs[c][tid]; s += v; s2 += v * v; }
        float mu = s * (1.f / DIM);
        float va = s2 * (1.f / DIM) - mu * mu;
        float rs = rsqrtf(va + 1e-5f);
        bf16* yp = ypk + tid * 72;
#pragma unroll
        for (int c = 0; c < DIM; c++) yp[c] = f2bf((xs[c][tid] - mu) * rs * g[c] + bln[c]);
#pragma unroll
        for (int c = DIM; c < 64; c++) yp[c] = f2bf(0.f);
    }
    __syncthreads();
    int lane = tid & 31, colBase = (tid >> 5) * 16;
    int half = lane >> 4, col = lane & 15;
    bf16* op = out + (size_t)b * C3 * NPIX;
    for (int mt = 0; mt < C3 / 16; mt++) {
        v8f acc = {};
#pragma unroll
        for (int ks = 0; ks < 2; ks++) {
            v16bf a = load_a_frag(Wb, 64, mt, ks, lane);
            v16bf bb = load_b_frag(ypk, 72, colBase, ks, lane);
            acc = __builtin_amdgcn_wmma_f32_16x16x32_bf16(false, a, false, bb,
                                                          (short)0, acc, false, false);
        }
#pragma unroll
        for (int v = 0; v < 8; v++) {
            int row = mt * 16 + v + 8 * half;
            op[(size_t)row * NPIX + pix0 + colBase + col] = f2bf(acc[v] + bias[row]);
        }
    }
}

// ---- depthwise 3x3 (+bias, optional exact GELU) ---------------------------
template <bool GELU>
__global__ __launch_bounds__(256) void k_dw3(const bf16* __restrict__ in,
                                             const float* __restrict__ w,
                                             const float* __restrict__ bias,
                                             bf16* __restrict__ out, int CH) {
    int ch = blockIdx.z % CH;
    int px = blockIdx.x * 32 + threadIdx.x;
    int py = blockIdx.y * 8 + threadIdx.y;
    const bf16* ip = in + (size_t)blockIdx.z * NPIX;
    if (py + 9 < HH) __builtin_prefetch(ip + (py + 9) * WW + px, 0, 1);
    const float* wp = w + ch * 9;
    float acc = 0.f;
#pragma unroll
    for (int dy = -1; dy <= 1; dy++) {
        int yy = py + dy;
        if (yy < 0 || yy >= HH) continue;
#pragma unroll
        for (int dx = -1; dx <= 1; dx++) {
            int xx = px + dx;
            if (xx < 0 || xx >= WW) continue;
            acc += wp[(dy + 1) * 3 + (dx + 1)] * (float)ip[yy * WW + xx];
        }
    }
    acc += bias[ch];
    if (GELU) acc = 0.5f * acc * (1.f + erff(acc * 0.70710678118654752f));
    out[(size_t)blockIdx.z * NPIX + py * WW + px] = f2bf(acc);
}

// ---- Gram + norm reduction per (b,h): G[6][6], |q|^2[6], |k|^2[6] ---------
__global__ __launch_bounds__(256) void k_gram(const bf16* __restrict__ qkvd,
                                              float* __restrict__ acc) {
    int bh = blockIdx.y, b = bh >> 3, h = bh & 7;
    int slab = blockIdx.x, tid = threadIdx.x;
    const bf16* qp = qkvd + ((size_t)b * C3 + h * HC) * NPIX;
    const bf16* kp = qp + (size_t)DIM * NPIX;
    float a[48];
#pragma unroll
    for (int i = 0; i < 48; i++) a[i] = 0.f;
    for (int it = 0; it < 16; it++) {
        int p = slab * 4096 + it * 256 + tid;
        float qv[HC], kv[HC];
#pragma unroll
        for (int c = 0; c < HC; c++) {
            qv[c] = (float)qp[(size_t)c * NPIX + p];
            kv[c] = (float)kp[(size_t)c * NPIX + p];
        }
#pragma unroll
        for (int c = 0; c < HC; c++) {
#pragma unroll
            for (int d = 0; d < HC; d++) a[c * HC + d] += qv[c] * kv[d];
            a[36 + c] += qv[c] * qv[c];
            a[42 + c] += kv[c] * kv[c];
        }
    }
    int lane = tid & 31;
    float* ga = acc + bh * 48;
#pragma unroll
    for (int i = 0; i < 48; i++) {
        float v = a[i];
        v += __shfl_xor(v, 1);
        v += __shfl_xor(v, 2);
        v += __shfl_xor(v, 4);
        v += __shfl_xor(v, 8);
        v += __shfl_xor(v, 16);
        if (lane == 0) atomicAdd(ga + i, v);
    }
}

// ---- softmax + W_eff[b] = W_proj * blockdiag(attn_b) ----------------------
__global__ __launch_bounds__(64) void k_weff(const float* __restrict__ acc,
                                             const float* __restrict__ scale,
                                             const float* __restrict__ wproj,
                                             bf16* __restrict__ weff) {
    __shared__ float attn[NHEAD][HC][HC];
    int b = blockIdx.x, tid = threadIdx.x;
    if (tid < NHEAD * HC) {
        int h = tid / HC, c = tid % HC;
        const float* ga = acc + (b * NHEAD + h) * 48;
        float qn = fmaxf(sqrtf(ga[36 + c]), 1e-12f);
        float s[HC];
        float mx = -1e30f;
#pragma unroll
        for (int d = 0; d < HC; d++) {
            float kn = fmaxf(sqrtf(ga[42 + d]), 1e-12f);
            s[d] = ga[c * HC + d] / (qn * kn) * scale[h];
            mx = fmaxf(mx, s[d]);
        }
        float sum = 0.f;
#pragma unroll
        for (int d = 0; d < HC; d++) { s[d] = expf(s[d] - mx); sum += s[d]; }
        float inv = 1.f / sum;
#pragma unroll
        for (int d = 0; d < HC; d++) attn[h][c][d] = s[d] * inv;
    }
    __syncthreads();
    bf16* wb = weff + (size_t)b * DIM * 64;
    for (int e = tid; e < DIM * 64; e += 64) {
        int o = e >> 6, j = e & 63;
        float v = 0.f;
        if (j < DIM) {
            int h = j / HC, d = j % HC;
#pragma unroll
            for (int c = 0; c < HC; c++) v += wproj[o * DIM + h * HC + c] * attn[h][c][d];
        }
        wb[e] = f2bf(v);
    }
}

// ---- x + W_eff@v_dw + b_proj, fused LN2 -> xmid(fp32), y2(bf16) -----------
__global__ __launch_bounds__(256) void k_attnproj(const bf16* __restrict__ qkvd,
                                                  const bf16* __restrict__ weff,
                                                  const float* __restrict__ x,
                                                  const float* __restrict__ bproj,
                                                  const float* __restrict__ g2,
                                                  const float* __restrict__ b2,
                                                  float* __restrict__ xmid,
                                                  bf16* __restrict__ y2) {
    __shared__ __align__(32) bf16 pk[TN * 72];
    int tid = threadIdx.x;
    int pix0 = blockIdx.x * TN;
    int b = blockIdx.y;
    const bf16* vp = qkvd + ((size_t)b * C3 + 2 * DIM) * NPIX;
    for (int e = tid; e < DIM * TN; e += 256) {
        int r = e >> 7, i = e & 127;
        pk[i * 72 + r] = vp[(size_t)r * NPIX + pix0 + i];
    }
    for (int e = tid; e < 16 * TN; e += 256) {
        int r = 48 + (e >> 7), i = e & 127;
        pk[i * 72 + r] = f2bf(0.f);
    }
    __syncthreads();
    int lane = tid & 31, colBase = (tid >> 5) * 16;
    int half = lane >> 4, col = lane & 15;
    const bf16* Wb = weff + (size_t)b * DIM * 64;
    const float* xb = x + (size_t)b * DIM * NPIX;
    float vals[3][8];
    for (int mt = 0; mt < 3; mt++) {
        v8f acc = {};
#pragma unroll
        for (int ks = 0; ks < 2; ks++) {
            v16bf a = load_a_frag(Wb, 64, mt, ks, lane);
            v16bf bb = load_b_frag(pk, 72, colBase, ks, lane);
            acc = __builtin_amdgcn_wmma_f32_16x16x32_bf16(false, a, false, bb,
                                                          (short)0, acc, false, false);
        }
#pragma unroll
        for (int v = 0; v < 8; v++) {
            int row = mt * 16 + v + 8 * half;
            vals[mt][v] = acc[v] + bproj[row] + xb[(size_t)row * NPIX + pix0 + colBase + col];
        }
    }
    // LN2: lanes L and L^16 hold the same pixel column (C-fragment layout)
    float s = 0.f, s2 = 0.f;
#pragma unroll
    for (int mt = 0; mt < 3; mt++)
#pragma unroll
        for (int v = 0; v < 8; v++) { float t = vals[mt][v]; s += t; s2 += t * t; }
    s += __shfl_xor(s, 16);
    s2 += __shfl_xor(s2, 16);
    float mu = s * (1.f / DIM);
    float va = s2 * (1.f / DIM) - mu * mu;
    float rs = rsqrtf(va + 1e-5f);
    float* xmb = xmid + (size_t)b * DIM * NPIX;
    bf16* y2b = y2 + (size_t)b * DIM * NPIX;
#pragma unroll
    for (int mt = 0; mt < 3; mt++)
#pragma unroll
        for (int v = 0; v < 8; v++) {
            int row = mt * 16 + v + 8 * half;
            size_t idx = (size_t)row * NPIX + pix0 + colBase + col;
            float t = vals[mt][v];
            xmb[idx] = t;
            y2b[idx] = f2bf((t - mu) * rs * g2[row] + b2[row]);
        }
}

// ---- FFN1: hid = W_f1(192x48) @ y2 + b_f1 ---------------------------------
__global__ __launch_bounds__(256) void k_ffn1(const bf16* __restrict__ y2,
                                              const bf16* __restrict__ Wb,
                                              const float* __restrict__ bias,
                                              bf16* __restrict__ hid) {
    __shared__ __align__(32) bf16 pk[TN * 72];
    int tid = threadIdx.x;
    int pix0 = blockIdx.x * TN;
    int b = blockIdx.y;
    const bf16* yp = y2 + (size_t)b * DIM * NPIX;
    for (int e = tid; e < DIM * TN; e += 256) {
        int r = e >> 7, i = e & 127;
        pk[i * 72 + r] = yp[(size_t)r * NPIX + pix0 + i];
    }
    for (int e = tid; e < 16 * TN; e += 256) {
        int r = 48 + (e >> 7), i = e & 127;
        pk[i * 72 + r] = f2bf(0.f);
    }
    __syncthreads();
    int lane = tid & 31, colBase = (tid >> 5) * 16;
    int half = lane >> 4, col = lane & 15;
    bf16* op = hid + (size_t)b * HIDN * NPIX;
    for (int mt = 0; mt < HIDN / 16; mt++) {
        v8f acc = {};
#pragma unroll
        for (int ks = 0; ks < 2; ks++) {
            v16bf a = load_a_frag(Wb, 64, mt, ks, lane);
            v16bf bb = load_b_frag(pk, 72, colBase, ks, lane);
            acc = __builtin_amdgcn_wmma_f32_16x16x32_bf16(false, a, false, bb,
                                                          (short)0, acc, false, false);
        }
#pragma unroll
        for (int v = 0; v < 8; v++) {
            int row = mt * 16 + v + 8 * half;
            op[(size_t)row * NPIX + pix0 + colBase + col] = f2bf(acc[v] + bias[row]);
        }
    }
}

// ---- FFN2: out = xmid + W_f2(48x192) @ hidd + b_f2 ------------------------
__global__ __launch_bounds__(256) void k_ffn2(const bf16* __restrict__ hd,
                                              const bf16* __restrict__ Wb,
                                              const float* __restrict__ bias,
                                              const float* __restrict__ xmid,
                                              float* __restrict__ outp) {
    __shared__ __align__(32) bf16 pk[TN * 200];
    int tid = threadIdx.x;
    int pix0 = blockIdx.x * TN;
    int b = blockIdx.y;
    const bf16* hp = hd + (size_t)b * HIDN * NPIX;
    for (int e = tid; e < HIDN * TN; e += 256) {
        int r = e >> 7, i = e & 127;
        pk[i * 200 + r] = hp[(size_t)r * NPIX + pix0 + i];
    }
    __syncthreads();
    int lane = tid & 31, colBase = (tid >> 5) * 16;
    int half = lane >> 4, col = lane & 15;
    const float* xmb = xmid + (size_t)b * DIM * NPIX;
    float* op = outp + (size_t)b * DIM * NPIX;
    for (int mt = 0; mt < 3; mt++) {
        v8f acc = {};
#pragma unroll
        for (int ks = 0; ks < 6; ks++) {
            v16bf a = load_a_frag(Wb, HIDN, mt, ks, lane);
            v16bf bb = load_b_frag(pk, 200, colBase, ks, lane);
            acc = __builtin_amdgcn_wmma_f32_16x16x32_bf16(false, a, false, bb,
                                                          (short)0, acc, false, false);
        }
#pragma unroll
        for (int v = 0; v < 8; v++) {
            int row = mt * 16 + v + 8 * half;
            size_t idx = (size_t)row * NPIX + pix0 + colBase + col;
            op[idx] = acc[v] + bias[row] + xmb[idx];
        }
    }
}

extern "C" void kernel_launch(void* const* d_in, const int* in_sizes, int n_in,
                              void* d_out, int out_size, void* d_ws, size_t ws_size,
                              hipStream_t stream) {
    (void)in_sizes; (void)n_in; (void)out_size; (void)ws_size;
    const float* x      = (const float*)d_in[0];
    const float* ln1_g  = (const float*)d_in[1];
    const float* ln1_b  = (const float*)d_in[2];
    const float* w_qkv  = (const float*)d_in[3];
    const float* b_qkv  = (const float*)d_in[4];
    const float* w_dw   = (const float*)d_in[5];
    const float* b_dw   = (const float*)d_in[6];
    const float* scale  = (const float*)d_in[7];
    const float* w_proj = (const float*)d_in[8];
    const float* b_proj = (const float*)d_in[9];
    const float* ln2_g  = (const float*)d_in[10];
    const float* ln2_b  = (const float*)d_in[11];
    const float* w_f1   = (const float*)d_in[12];
    const float* b_f1   = (const float*)d_in[13];
    const float* w_fdw  = (const float*)d_in[14];
    const float* b_fdw  = (const float*)d_in[15];
    const float* w_f2   = (const float*)d_in[16];
    const float* b_f2   = (const float*)d_in[17];

    char* ws = (char*)d_ws;
    // region0 aliases {qkv, qkvd} (phase A) with {hid, hidd} (phase B)
    const size_t qkvB  = (size_t)BATCH * C3 * NPIX * 2;    // 75,497,472
    const size_t hidB  = (size_t)BATCH * HIDN * NPIX * 2;  // 100,663,296
    const size_t r0B   = 2 * hidB;                         // 201,326,592
    size_t off = 0;
    bf16* qkv  = (bf16*)(ws + off);
    bf16* qkvd = (bf16*)(ws + off + qkvB);
    bf16* hid  = (bf16*)(ws + off);
    bf16* hidd = (bf16*)(ws + off + hidB);
    off += r0B;
    float* xmid = (float*)(ws + off); off += (size_t)BATCH * DIM * NPIX * 4;
    bf16*  y2   = (bf16*) (ws + off); off += (size_t)BATCH * DIM * NPIX * 2;
    float* acc  = (float*)(ws + off); off += (size_t)BATCH * NHEAD * 48 * 4 + 256;
    bf16* weffb = (bf16*) (ws + off); off += (size_t)BATCH * DIM * 64 * 2;
    bf16* wqkvb = (bf16*) (ws + off); off += (size_t)C3 * 64 * 2;
    bf16* wf1b  = (bf16*) (ws + off); off += (size_t)HIDN * 64 * 2;
    bf16* wf2b  = (bf16*) (ws + off); off += (size_t)DIM * HIDN * 2;

    hipMemsetAsync(acc, 0, (size_t)BATCH * NHEAD * 48 * 4, stream);
    k_prep<<<120, 256, 0, stream>>>(w_qkv, w_f1, w_f2, wqkvb, wf1b, wf2b);

    dim3 gGemm(NPIX / TN, BATCH);
    k_qkv<<<gGemm, 256, 0, stream>>>(x, ln1_g, ln1_b, wqkvb, b_qkv, qkv);
    k_dw3<false><<<dim3(WW / 32, HH / 8, BATCH * C3), dim3(32, 8), 0, stream>>>(
        qkv, w_dw, b_dw, qkvd, C3);
    k_gram<<<dim3(16, BATCH * NHEAD), 256, 0, stream>>>(qkvd, acc);
    k_weff<<<BATCH, 64, 0, stream>>>(acc, scale, w_proj, weffb);
    k_attnproj<<<gGemm, 256, 0, stream>>>(qkvd, weffb, x, b_proj, ln2_g, ln2_b, xmid, y2);
    k_ffn1<<<gGemm, 256, 0, stream>>>(y2, wf1b, b_f1, hid);
    k_dw3<true><<<dim3(WW / 32, HH / 8, BATCH * HIDN), dim3(32, 8), 0, stream>>>(
        hid, w_fdw, b_fdw, hidd, HIDN);
    k_ffn2<<<gGemm, 256, 0, stream>>>(hidd, wf2b, b_f2, xmid, (float*)d_out);
}
